// TernarySwiGLU_75067438400005
// MI455X (gfx1250) — matrix operand order repaired and checked by reading, not compiled, not run on hardware
//
#include <hip/hip_runtime.h>
#include <math.h>

typedef __attribute__((ext_vector_type(16))) _Float16 v16h;
typedef __attribute__((ext_vector_type(8)))  float    v8f;

enum { MODE_SWIGLU = 0, MODE_GELU = 1, MODE_FINAL = 2 };

union FragU { v16h v; uint4 q[2]; };
union PK4  { _Float16 h[4]; uint2 u; };

static __device__ __forceinline__ float silu_f(float g) {
  return g / (1.0f + __expf(-g));
}
static __device__ __forceinline__ float gelu_f(float g) {
  return 0.5f * g * (1.0f + erff(g * 0.70710678118654752f));
}

// C[m,n] = sum_k A[m,k] * W[n,k]  (A row-major MxK, W row-major NxK == B^T)
// A is f32 (layer 1) or f16 (later layers); W is f32 ternary, converted to f16
// while staging to LDS. Scales applied in the epilogue (f32).
template<int MODE, bool A_F32, int WTN, int NB>
__global__ __launch_bounds__(256) void ternary_gemm(
    const void* __restrict__ Ain,
    const float* __restrict__ W1, const float* __restrict__ S1,
    const float* __restrict__ W2, const float* __restrict__ S2,
    void* __restrict__ Out, int M, int N, int K)
{
  constexpr int WTM  = 4;        // 4 M-tiles of 16 per wave (wave grid 2x4)
  constexpr int BM   = 128;
  constexpr int LDSK = 40;       // 32 + 8 halves pad (80B row, 16B aligned)
  constexpr int NB2  = (MODE == MODE_SWIGLU) ? NB : 1;
  constexpr int WCNT = (NB * 8) / 256;   // float4 chunks/thread per weight tile

  __shared__ unsigned short As [2][BM  * LDSK];
  __shared__ unsigned short Bs1[2][NB  * LDSK];
  __shared__ unsigned short Bs2[2][NB2 * LDSK];

  const int tid    = threadIdx.x;
  const int lane   = tid & 31;
  const int l15    = lane & 15;
  const int hb     = lane >> 4;           // lane half (ISA fragment striping)
  const int waveId = tid >> 5;
  const int wmBase = (waveId >> 2) * (WTM * 16);
  const int wnBase = (waveId & 3)  * (WTN * 16);

  // group-of-8 block swizzle along M for L2 weight reuse
  const int ntiles = N / NB;
  const int mtiles = M / BM;
  const int GRP = 8;
  int bid = blockIdx.x;
  int grp = bid / (GRP * ntiles);
  int rem = bid - grp * (GRP * ntiles);
  int gm  = grp * GRP;
  int gsz = (mtiles - gm) < GRP ? (mtiles - gm) : GRP;
  const int mi = gm + rem % gsz;
  const int ni = rem / gsz;

  const int nk = K / 32;

  float4 aF[A_F32 ? 4 : 1];
  uint4  aH[A_F32 ? 1 : 2];
  float4 w1r[WCNT];
  float4 w2r[(MODE == MODE_SWIGLU) ? WCNT : 1];

  const float*          Afp = (const float*)Ain          + (size_t)mi * BM * K;
  const unsigned short* Ahp = (const unsigned short*)Ain + (size_t)mi * BM * K;
  const float* W1p = W1 + (size_t)ni * NB * K;
  const float* W2p = (MODE == MODE_SWIGLU) ? (W2 + (size_t)ni * NB * K)
                                           : (const float*)nullptr;

  auto loadA = [&](int kt) {
    if constexpr (A_F32) {
      #pragma unroll
      for (int i = 0; i < 4; ++i) {
        int idx = tid + i * 256, row = idx >> 3, c4 = idx & 7;
        aF[i] = *(const float4*)(Afp + (size_t)row * K + kt * 32 + c4 * 4);
      }
    } else {
      #pragma unroll
      for (int i = 0; i < 2; ++i) {
        int idx = tid + i * 256, row = idx >> 2, c8 = idx & 3;
        aH[i] = *(const uint4*)(Ahp + (size_t)row * K + kt * 32 + c8 * 8);
      }
    }
  };
  auto loadW = [&](int kt) {
    #pragma unroll
    for (int i = 0; i < WCNT; ++i) {
      int idx = tid + i * 256, row = idx >> 3, c4 = idx & 7;
      w1r[i] = *(const float4*)(W1p + (size_t)row * K + kt * 32 + c4 * 4);
      if constexpr (MODE == MODE_SWIGLU)
        w2r[i] = *(const float4*)(W2p + (size_t)row * K + kt * 32 + c4 * 4);
    }
  };
  auto cvtStore = [&](unsigned short* dst, const float4& f) {
    PK4 p;
    p.h[0] = (_Float16)f.x; p.h[1] = (_Float16)f.y;
    p.h[2] = (_Float16)f.z; p.h[3] = (_Float16)f.w;
    *(uint2*)dst = p.u;
  };
  auto storeA = [&](int buf) {
    if constexpr (A_F32) {
      #pragma unroll
      for (int i = 0; i < 4; ++i) {
        int idx = tid + i * 256, row = idx >> 3, c4 = idx & 7;
        cvtStore(&As[buf][row * LDSK + c4 * 4], aF[i]);
      }
    } else {
      #pragma unroll
      for (int i = 0; i < 2; ++i) {
        int idx = tid + i * 256, row = idx >> 2, c8 = idx & 3;
        *(uint4*)&As[buf][row * LDSK + c8 * 8] = aH[i];
      }
    }
  };
  auto storeW = [&](int buf) {
    #pragma unroll
    for (int i = 0; i < WCNT; ++i) {
      int idx = tid + i * 256, row = idx >> 3, c4 = idx & 7;
      cvtStore(&Bs1[buf][row * LDSK + c4 * 4], w1r[i]);
      if constexpr (MODE == MODE_SWIGLU)
        cvtStore(&Bs2[buf][row * LDSK + c4 * 4], w2r[i]);
    }
  };

  v8f acc1[WTM][WTN];
  v8f acc2[(MODE == MODE_SWIGLU) ? WTM : 1][WTN];
  v8f vzero = {};
  #pragma unroll
  for (int m = 0; m < WTM; ++m)
    #pragma unroll
    for (int n = 0; n < WTN; ++n) {
      acc1[m][n] = vzero;
      if constexpr (MODE == MODE_SWIGLU) acc2[m][n] = vzero;
    }

  loadA(0); loadW(0);
  storeA(0); storeW(0);
  __syncthreads();

  for (int kt = 0; kt < nk; ++kt) {
    const int  cur  = kt & 1;
    const bool more = (kt + 1) < nk;
    if (more) { loadA(kt + 1); loadW(kt + 1); }   // global latency over compute

    // Fragments in contiguous k order (k = 16*hb + s), same for A and B:
    // A 16x32 f16: lanes<16 hold K 0..15; lanes>=16 hold K 16..31.
    // B 32x16 f16: lane%16 = N col; lanes<16 K=0..15, lanes>=16 K=16..31.
    FragU a[WTM], b1[WTN], b2[(MODE == MODE_SWIGLU) ? WTN : 1];
    const int akb = hb * 16;
    const int bkb = hb * 16;
    #pragma unroll
    for (int m = 0; m < WTM; ++m) {
      int arow = wmBase + m * 16 + l15;
      a[m].q[0] = *(const uint4*)&As[cur][arow * LDSK + akb];
      a[m].q[1] = *(const uint4*)&As[cur][arow * LDSK + akb + 8];
    }
    #pragma unroll
    for (int n = 0; n < WTN; ++n) {
      int brow = wnBase + n * 16 + l15;
      b1[n].q[0] = *(const uint4*)&Bs1[cur][brow * LDSK + bkb];
      b1[n].q[1] = *(const uint4*)&Bs1[cur][brow * LDSK + bkb + 8];
      if constexpr (MODE == MODE_SWIGLU) {
        b2[n].q[0] = *(const uint4*)&Bs2[cur][brow * LDSK + bkb];
        b2[n].q[1] = *(const uint4*)&Bs2[cur][brow * LDSK + bkb + 8];
      }
    }
    #pragma unroll
    for (int m = 0; m < WTM; ++m)
      #pragma unroll
      for (int n = 0; n < WTN; ++n) {
        acc1[m][n] = __builtin_amdgcn_wmma_f32_16x16x32_f16(
            false, a[m].v, false, b1[n].v, (short)0, acc1[m][n], false, false);
        if constexpr (MODE == MODE_SWIGLU)
          acc2[m][n] = __builtin_amdgcn_wmma_f32_16x16x32_f16(
              false, a[m].v, false, b2[n].v, (short)0, acc2[m][n], false, false);
      }

    if (more) {
      storeA(cur ^ 1); storeW(cur ^ 1);   // other buffer: safe, one barrier/iter
      __syncthreads();
    }
  }

  // epilogue: C/D layout -> VGPR r: M=r (lanes<16) or M=8+r (lanes>=16), N=lane%16
  #pragma unroll
  for (int n = 0; n < WTN; ++n) {
    const int col = ni * NB + wnBase + n * 16 + l15;
    const float s1 = S1[col];
    float s2 = 0.0f;
    if constexpr (MODE == MODE_SWIGLU) s2 = S2[col];
    #pragma unroll
    for (int m = 0; m < WTM; ++m) {
      #pragma unroll
      for (int r = 0; r < 8; ++r) {
        const size_t row = (size_t)mi * BM + wmBase + m * 16 + hb * 8 + r;
        const float g = acc1[m][n][r] * s1;
        if constexpr (MODE == MODE_SWIGLU) {
          const float h = silu_f(g) * (acc2[m][n][r] * s2);
          ((_Float16*)Out)[row * (size_t)N + col] = (_Float16)h;
        } else if constexpr (MODE == MODE_GELU) {
          ((_Float16*)Out)[row * (size_t)N + col] = (_Float16)gelu_f(g);
        } else {
          ((float*)Out)[row * (size_t)N + col] = g;
        }
      }
    }
  }
}

extern "C" void kernel_launch(void* const* d_in, const int* in_sizes, int n_in,
                              void* d_out, int out_size, void* d_ws, size_t ws_size,
                              hipStream_t stream) {
  const float* x    = (const float*)d_in[0];
  const float* w1_w = (const float*)d_in[1];
  const float* w1_s = (const float*)d_in[2];
  const float* w2_w = (const float*)d_in[3];
  const float* w2_s = (const float*)d_in[4];
  const float* h1_w = (const float*)d_in[5];
  const float* h1_s = (const float*)d_in[6];
  const float* h2_w = (const float*)d_in[7];
  const float* h2_s = (const float*)d_in[8];
  const float* w3_w = (const float*)d_in[9];
  const float* w3_s = (const float*)d_in[10];

  const int M = 4 * 2048;       // 8192 rows
  const int D_IN = 1024, D_HID = 4096;

  _Float16* hA = (_Float16*)d_ws;                       // 8192*4096 f16 (64 MiB)
  _Float16* hB = hA + (size_t)M * D_HID;                // 8192*4096 f16 (64 MiB)

  dim3 blk(256);

  // L1: fused SwiGLU pair  (A=f32 x, BN=64, dual accumulators)
  ternary_gemm<MODE_SWIGLU, true, 1, 64>
      <<<dim3((M / 128) * (D_HID / 64)), blk, 0, stream>>>(
          x, w1_w, w1_s, w2_w, w2_s, hA, M, D_HID, D_IN);

  // L2: gelu(h @ h1^T * s)
  ternary_gemm<MODE_GELU, false, 2, 128>
      <<<dim3((M / 128) * (D_HID / 128)), blk, 0, stream>>>(
          hA, h1_w, h1_s, nullptr, nullptr, hB, M, D_HID, D_HID);

  // L3: gelu(h @ h2^T * s)
  ternary_gemm<MODE_GELU, false, 2, 128>
      <<<dim3((M / 128) * (D_HID / 128)), blk, 0, stream>>>(
          hB, h2_w, h2_s, nullptr, nullptr, hA, M, D_HID, D_HID);

  // L4: out = h @ w3^T * s  (f32 output)
  ternary_gemm<MODE_FINAL, false, 2, 128>
      <<<dim3((M / 128) * (D_IN / 128)), blk, 0, stream>>>(
          hA, w3_w, w3_s, nullptr, nullptr, d_out, M, D_IN, D_HID);
}
